// MultiHeadSelfAttention_9466107920542
// MI455X (gfx1250) — matrix operand order repaired and checked by reading, not compiled, not run on hardware
//
#include <hip/hip_runtime.h>

// ---------------------------------------------------------------------------
// MHSA for B=2, S=2048, D=512, H=8 on gfx1250 (CDNA5, wave32, WMMA).
// All matmuls use v_wmma_f32_16x16x32_bf16 (bf16 A/B, f32 accumulate).
// Attention uses 64-query tiles so all 8 waves stay WMMA-busy in both the
// QK^T and PV phases; k/v tiles stream via async global->LDS when available.
// ---------------------------------------------------------------------------

#define DEV __device__ __forceinline__

typedef __attribute__((ext_vector_type(16))) __bf16     bf16x16;
typedef __attribute__((ext_vector_type(8)))  float      f32x8;
typedef __attribute__((ext_vector_type(4)))  unsigned   u32x4;

constexpr int BB = 2, SS = 2048, DD = 512, HH = 8;

// ---- optional CDNA5 async global->LDS copy path (ASYNCcnt tracked) --------
#if defined(__AMDGCN__) && \
    __has_builtin(__builtin_amdgcn_global_load_async_to_lds_b128) && \
    __has_builtin(__builtin_amdgcn_s_wait_asynccnt)
#define USE_ASYNC_COPY 1
#else
#define USE_ASYNC_COPY 0
#endif

#if USE_ASYNC_COPY
// Builtin signature (from hipcc diagnostic): 16-byte int-vector pointers,
// src in addrspace(1) (global), dst in addrspace(3) (LDS), imm offset, imm cpol.
typedef int v4i_vs __attribute__((vector_size(16)));
typedef __attribute__((address_space(1))) v4i_vs GlobV4;
typedef __attribute__((address_space(3))) v4i_vs LdsV4;
// generic->AS1: flat pointers to global memory are bit-identical to AS1.
// generic->AS3: flat LDS pointers carry the LDS byte offset in the low 32 bits.
DEV void async_cp16(const void* g, void* l) {
  __builtin_amdgcn_global_load_async_to_lds_b128(
      (GlobV4*)(unsigned long long)(uintptr_t)g,
      (LdsV4*)(unsigned)(uintptr_t)l, 0, 0);
}
DEV void async_wait0() { __builtin_amdgcn_s_wait_asynccnt(0); }
#else
DEV void async_cp16(const void* g, void* l) {
  *(u32x4*)l = *(const u32x4*)g;
}
DEV void async_wait0() {}
#endif

union FragU { bf16x16 v; u32x4 q[2]; };

// A-matrix 16x32 bf16 fragment (ISA 7.12.2): lane holds row (lane&15);
// lanes 0-15 carry K = {0..7, 16..23}+kOff, lanes 16-31 K = {8..15, 24..31}+kOff.
// base is row-major [row][K], `stride` elements per row. Two 16B LDS loads.
DEV bf16x16 lds_frag_a(const __bf16* base, int row0, int stride, int kOff, int lane) {
  const __bf16* p = base + (size_t)(row0 + (lane & 15)) * stride + kOff + ((lane >> 4) << 3);
  FragU f;
  f.q[0] = *(const u32x4*)(p);
  f.q[1] = *(const u32x4*)(p + 16);
  return f.v;
}

// B-matrix 32x16 bf16 fragment: lane holds column (lane&15); lanes 0-15 carry
// K = {0..7, 16..23}, lanes 16-31 K = {8..15, 24..31} (matches A). baseT is stored
// column-major: [col][K] with `stride` elements per column.
DEV bf16x16 lds_frag_b(const __bf16* baseT, int col0, int stride, int kOff, int lane) {
  const __bf16* p = baseT + (size_t)(col0 + (lane & 15)) * stride + kOff + ((lane >> 4) << 3);
  FragU f;
  f.q[0] = *(const u32x4*)(p);
  f.q[1] = *(const u32x4*)(p + 16);
  return f.v;
}

DEV f32x8 wmma_bf16(bf16x16 a, bf16x16 b, f32x8 c) {
  return __builtin_amdgcn_wmma_f32_16x16x32_bf16(
      /*neg_a=*/false, a, /*neg_b=*/false, b,
      /*c_mod=*/(short)0, c, /*reuse_a=*/false, /*reuse_b=*/false);
}

// ---------------------------------------------------------------------------
// GEMM: out[M,N] = A[M,K] * W[K,N] + bias[N]
//   A_BF16     : A is bf16 (else f32, converted on the fly)
//   OUT_F32    : write f32 (else bf16)
//   TRANS_OUT  : write out[N,M] instead of out[M,N] (used for V^T)
//   QKV_BATCH  : blockIdx.z = b*H + h; A indexed by b, W/bias by h, out by z
// Block = 256 threads (8 waves). Tile 128(M) x 128(N), K-step 32.
// Waves: 4 in M (32 rows each) x 2 in N (64 cols each) -> 8 WMMA accs/wave.
// ---------------------------------------------------------------------------
template<bool A_BF16, bool OUT_F32, bool TRANS_OUT, bool QKV_BATCH>
__global__ __launch_bounds__(256)
void gemm_bias_wmma(const void* __restrict__ A_, const float* __restrict__ W,
                    const float* __restrict__ bias, void* __restrict__ out_,
                    int M, int N, int K) {
  __shared__ __bf16 As[128 * 32];   // [row][k]        8 KB
  __shared__ __bf16 Bs[128 * 32];   // [col][k] (W^T)  8 KB

  const char* Ab = (const char*)A_;
  char* outb = (char*)out_;
  if (QKV_BATCH) {
    const int z = blockIdx.z;
    const int b = z / HH, h = z % HH;
    Ab   += (size_t)b * M * K * sizeof(float);
    W    += (size_t)h * K * N;
    bias += (size_t)h * N;
    outb += (size_t)z * M * N * sizeof(__bf16);
  }

  const int tid   = threadIdx.x;
  const int lane  = tid & 31;
  const int wave  = tid >> 5;
  const int wm    = wave >> 1;            // 0..3
  const int wn    = wave & 1;             // 0..1
  const int tileM = blockIdx.x * 128;
  const int tileN = blockIdx.y * 128;

  f32x8 acc0[4], acc1[4];
#pragma unroll
  for (int n = 0; n < 4; ++n)
#pragma unroll
    for (int i = 0; i < 8; ++i) { acc0[n][i] = 0.f; acc1[n][i] = 0.f; }

  for (int kt = 0; kt < K; kt += 32) {
    __syncthreads();
    { // stage A tile: each thread fills 16 contiguous K of one row
      const int r  = tid >> 1;            // 0..127
      const int k0 = (tid & 1) << 4;      // 0 | 16
      if (A_BF16) {
        const __bf16* src = (const __bf16*)Ab + (size_t)(tileM + r) * K + kt + k0;
        __bf16* dst = As + r * 32 + k0;
        async_cp16(src, dst);
        async_cp16(src + 8, dst + 8);
      } else {
        const float* src = (const float*)Ab + (size_t)(tileM + r) * K + kt + k0;
#pragma unroll
        for (int j = 0; j < 16; ++j) As[r * 32 + k0 + j] = (__bf16)src[j];
      }
    }
    { // stage W tile transposed into [col][k]
      const int kk = tid >> 3;            // 0..31
      const int n0 = (tid & 7) << 4;      // 0..112
      const float* src = W + (size_t)(kt + kk) * N + tileN + n0;
#pragma unroll
      for (int j = 0; j < 16; ++j) Bs[(n0 + j) * 32 + kk] = (__bf16)src[j];
      if (kt + 32 < K) __builtin_prefetch(src + (size_t)32 * N, 0, 1);
    }
    if (A_BF16) async_wait0();
    __syncthreads();

    const bf16x16 a0 = lds_frag_a(As, wm * 32 + 0,  32, 0, lane);
    const bf16x16 a1 = lds_frag_a(As, wm * 32 + 16, 32, 0, lane);
#pragma unroll
    for (int nf = 0; nf < 4; ++nf) {
      const bf16x16 bfr = lds_frag_b(Bs, wn * 64 + nf * 16, 32, 0, lane);
      acc0[nf] = wmma_bf16(a0, bfr, acc0[nf]);
      acc1[nf] = wmma_bf16(a1, bfr, acc1[nf]);
    }
  }

  // epilogue: C/D layout -> element i at row i+8*(lane>>4), col lane&15
#pragma unroll
  for (int mf = 0; mf < 2; ++mf) {
    f32x8* accs = mf ? acc1 : acc0;
    const int rbase = tileM + wm * 32 + mf * 16 + ((lane >> 4) << 3);
#pragma unroll
    for (int nf = 0; nf < 4; ++nf) {
      const int col = tileN + wn * 64 + nf * 16 + (lane & 15);
      const float bv = bias[col];
#pragma unroll
      for (int i = 0; i < 8; ++i) {
        const float v = accs[nf][i] + bv;
        if (OUT_F32)        ((float*)outb)[(size_t)(rbase + i) * N + col] = v;
        else if (TRANS_OUT) ((__bf16*)outb)[(size_t)col * M + (rbase + i)] = (__bf16)v;
        else                ((__bf16*)outb)[(size_t)(rbase + i) * N + col] = (__bf16)v;
      }
    }
  }
}

// ---------------------------------------------------------------------------
// Flash attention: grid = (S/64 query blocks, B*H). Block = 256 threads.
// q,k: bf16 [B,H,S,D]; vt: bf16 [B,H,D,S]; cat out: bf16 [B,S,H*D].
// Per key block (32 keys):
//   scores : 64q x 32k, waves split 4(M)x2(keys) -> 16 WMMA each (all 8 busy)
//   softmax: threads 0-63, one query row each (online, f32)
//   PV     : O[64 x 512], waves split 4(M rows of 16) x 2(N cols of 256)
//            -> 16 accumulators (128 VGPRs) and 16 WMMA per wave per block.
// ---------------------------------------------------------------------------
constexpr int ATTN_SMEM =
    64 * 512 * 2 +   // qs
    32 * 512 * 2 +   // ks
    512 * 32 * 2 +   // vs (v^T tile: [d][key])
    64 * 32 * 4 +    // Ss f32 scores
    64 * 32 * 2 +    // Ps bf16 probs
    3 * 64 * 4;      // mrow, lrow, cf   => 144128 bytes

__global__ __launch_bounds__(256)
void mhsa_flash_attn_wmma(const void* __restrict__ q_, const void* __restrict__ k_,
                          const void* __restrict__ vt_, void* __restrict__ cat_) {
  extern __shared__ char smem[];
  __bf16* qs   = (__bf16*)smem;                 // [64][512]
  __bf16* ks   = qs + 64 * 512;                 // [32][512]
  __bf16* vs   = ks + 32 * 512;                 // [512][32]
  float*  Ss   = (float*)(vs + 512 * 32);       // [64][32]
  __bf16* Ps   = (__bf16*)(Ss + 64 * 32);       // [64][32]
  float*  mrow = (float*)(Ps + 64 * 32);        // [64]
  float*  lrow = mrow + 64;                     // [64]
  float*  cf   = lrow + 64;                     // [64]

  const int qb0 = blockIdx.x * 64;
  const int bh  = blockIdx.y;
  const int b   = bh / HH, h = bh % HH;
  const __bf16* qh = (const __bf16*)q_  + (size_t)bh * SS * DD;
  const __bf16* kh = (const __bf16*)k_  + (size_t)bh * SS * DD;
  const __bf16* vh = (const __bf16*)vt_ + (size_t)bh * DD * SS;
  __bf16* cat = (__bf16*)cat_;

  const int tid  = threadIdx.x;
  const int lane = tid & 31;
  const int wave = tid >> 5;
  const int wmO  = wave >> 1;   // 0..3 : 16 query rows (scores + PV split)
  const int wnS  = wave & 1;    // 0..1 : 16 keys (scores) / 256 cols (PV)

  { // load q tile once: 64x512, 128 bf16 (16 x 16B chunks) per thread
    const int r  = tid >> 2;              // 0..63
    const int c0 = (tid & 3) * 128;
    const __bf16* src = qh + (size_t)(qb0 + r) * DD + c0;
    __bf16* dst = qs + r * 512 + c0;
#pragma unroll
    for (int i = 0; i < 16; ++i) async_cp16(src + i * 8, dst + i * 8);
  }
  if (tid < 64) { mrow[tid] = -1e30f; lrow[tid] = 0.f; }

  f32x8 acc[16];
#pragma unroll
  for (int n = 0; n < 16; ++n)
#pragma unroll
    for (int i = 0; i < 8; ++i) acc[n][i] = 0.f;

  for (int kb0 = 0; kb0 < SS; kb0 += 32) {
    __syncthreads();   // previous PV done before overwriting ks/vs
    { // k tile: [32 keys][512 d] straight copy (B-frag wants contiguous K=d)
      const int r  = tid >> 3;
      const int c0 = (tid & 7) * 64;
      const __bf16* src = kh + (size_t)(kb0 + r) * DD + c0;
      __bf16* dst = ks + r * 512 + c0;
#pragma unroll
      for (int i = 0; i < 8; ++i) async_cp16(src + i * 8, dst + i * 8);
    }
    { // v^T tile: [512 d][32 keys] straight copy from vt (keys contiguous)
      const int r0 = tid * 2;
#pragma unroll
      for (int rr = 0; rr < 2; ++rr) {
        const __bf16* src = vh + (size_t)(r0 + rr) * SS + kb0;
        __bf16* dst = vs + (r0 + rr) * 32;
#pragma unroll
        for (int i = 0; i < 4; ++i) async_cp16(src + i * 8, dst + i * 8);
      }
    }
    async_wait0();     // q (first iter) + k + v async copies landed in LDS
    __syncthreads();

    { // scores: each wave one 16x16 tile of S[64x32], contract over D=512
      f32x8 sacc;
#pragma unroll
      for (int i = 0; i < 8; ++i) sacc[i] = 0.f;
#pragma unroll
      for (int dc = 0; dc < 512; dc += 32) {
        const bf16x16 a   = lds_frag_a(qs, wmO * 16, 512, dc, lane);
        const bf16x16 bfr = lds_frag_b(ks, wnS * 16, 512, dc, lane);
        sacc = wmma_bf16(a, bfr, sacc);
      }
      const int col   = wnS * 16 + (lane & 15);
      const int rbase = wmO * 16 + ((lane >> 4) << 3);
#pragma unroll
      for (int i = 0; i < 8; ++i) Ss[(rbase + i) * 32 + col] = sacc[i];
    }
    __syncthreads();

    if (tid < 64) { // online softmax, one thread per query row
      const float mo = mrow[tid];
      float mx = mo;
      for (int j = 0; j < 32; ++j) mx = fmaxf(mx, Ss[tid * 32 + j]);
      const float c = __expf(mo - mx);
      float sum = 0.f;
      for (int j = 0; j < 32; ++j) {
        const float p = __expf(Ss[tid * 32 + j] - mx);
        sum += p;
        Ps[tid * 32 + j] = (__bf16)p;
      }
      mrow[tid] = mx;
      lrow[tid] = lrow[tid] * c + sum;
      cf[tid]   = c;
    }
    __syncthreads();

    { // rescale O then O += P V   (wave: rows wmO*16.., cols wnS*256..)
      const int rb = wmO * 16 + ((lane >> 4) << 3);
      float cv[8];
#pragma unroll
      for (int i = 0; i < 8; ++i) cv[i] = cf[rb + i];
#pragma unroll
      for (int nf = 0; nf < 16; ++nf)
#pragma unroll
        for (int i = 0; i < 8; ++i) acc[nf][i] *= cv[i];

      const bf16x16 a = lds_frag_a(Ps, wmO * 16, 32, 0, lane);
#pragma unroll
      for (int nf = 0; nf < 16; ++nf) {
        const bf16x16 bfr = lds_frag_b(vs, wnS * 256 + nf * 16, 32, 0, lane);
        acc[nf] = wmma_bf16(a, bfr, acc[nf]);
      }
    }
  }
  __syncthreads();

  { // finalize: O /= l, store to cat[b][s][h*D + col]
    const int rb = wmO * 16 + ((lane >> 4) << 3);
    float linv[8];
#pragma unroll
    for (int i = 0; i < 8; ++i) linv[i] = 1.f / lrow[rb + i];
#pragma unroll
    for (int nf = 0; nf < 16; ++nf) {
      const int col = wnS * 256 + nf * 16 + (lane & 15);
#pragma unroll
      for (int i = 0; i < 8; ++i) {
        const int s = qb0 + rb + i;
        cat[((size_t)(b * SS + s)) * (HH * DD) + h * DD + col] =
            (__bf16)(acc[nf][i] * linv[i]);
      }
    }
  }
}

// ---------------------------------------------------------------------------
extern "C" void kernel_launch(void* const* d_in, const int* in_sizes, int n_in,
                              void* d_out, int out_size, void* d_ws, size_t ws_size,
                              hipStream_t stream) {
  (void)in_sizes; (void)n_in; (void)out_size; (void)ws_size;
  const float* x  = (const float*)d_in[0];
  const float* Wq = (const float*)d_in[1];
  const float* bq = (const float*)d_in[2];
  const float* Wk = (const float*)d_in[3];
  const float* bk = (const float*)d_in[4];
  const float* Wv = (const float*)d_in[5];
  const float* bv = (const float*)d_in[6];
  const float* Wo = (const float*)d_in[7];
  const float* bo = (const float*)d_in[8];

  // workspace: qb | kb | vtb | cat, each B*H*S*D bf16 (33.5 MB) -> ~134 MB
  const size_t perMat = (size_t)BB * HH * SS * DD;
  char* ws  = (char*)d_ws;
  void* qb  = ws;
  void* kb  = ws + perMat * 2;
  void* vtb = ws + perMat * 4;
  void* cat = ws + perMat * 6;

  const dim3 blk(256);
  const dim3 gProj(SS / 128, DD / 128, BB * HH);
  gemm_bias_wmma<false, false, false, true><<<gProj, blk, 0, stream>>>(x, Wq, bq, qb,  SS, DD, DD);
  gemm_bias_wmma<false, false, false, true><<<gProj, blk, 0, stream>>>(x, Wk, bk, kb,  SS, DD, DD);
  gemm_bias_wmma<false, false, true,  true><<<gProj, blk, 0, stream>>>(x, Wv, bv, vtb, SS, DD, DD);

  (void)hipFuncSetAttribute((const void*)mhsa_flash_attn_wmma,
                            hipFuncAttributeMaxDynamicSharedMemorySize, ATTN_SMEM);
  const dim3 gAttn(SS / 64, BB * HH);
  mhsa_flash_attn_wmma<<<gAttn, blk, ATTN_SMEM, stream>>>(qb, kb, vtb, cat);

  const dim3 gOut((BB * SS) / 128, DD / 128, 1);
  gemm_bias_wmma<true, true, false, false><<<gOut, blk, 0, stream>>>(cat, Wo, bo, d_out,
                                                                     BB * SS, DD, HH * DD);
}
